// AlloGNN_77300821393453
// MI455X (gfx1250) — matrix operand order repaired and checked by reading, not compiled, not run on hardware
//
#include <hip/hip_runtime.h>
#include <hip/hip_bf16.h>
#include <math.h>

typedef __attribute__((ext_vector_type(16))) _Float16 v16h;
typedef __attribute__((ext_vector_type(8)))  float    v8f;

#define N_LINK 2000
#define N_FLOW 30000
#define N_PATH 50000
#define N_EDGE 120000
#define HIDN   256
#define NHEAD  4
#define HDFLAT 1024   // NHEAD * D (D == 256)

// ===========================================================================
// Fragment packing: fp32 row-major -> f16 hi/lo planes in WMMA fragment order.
// A plane layout: half APK[ (mt*KT + kt)*32 + lane ][16]   (v16h per lane)
// B plane layout: half BPK[ (nt*KT + kt)*32 + lane ][16]
// K is zero-padded to KT*32 here, so the GEMM loop needs no bounds checks.
// ===========================================================================
__global__ void pack_a_kernel(const float* __restrict__ A, int lda, int M, int K,
                              int KT, _Float16* __restrict__ APH,
                              _Float16* __restrict__ APL)
{
    int t = blockIdx.x * blockDim.x + threadIdx.x;          // one thread = one lane-frag
    int total = (M >> 4) * KT * 32;
    if (t >= total) return;
    const int lane = t & 31;
    const int rem  = t >> 5;
    const int kt   = rem % KT;
    const int mt   = rem / KT;
    const int m    = (mt << 4) + (lane & 15);
    const int kb   = (kt << 5) + ((lane >> 4) << 3);        // A: k sub-base 0/8
    const float* __restrict__ row = A + (size_t)m * lda;
    v16h hi, lo;
#pragma unroll
    for (int i = 0; i < 16; ++i) {
        int k = kb + (i & 7) + ((i >> 3) << 4);             // ISA 7.12.2 16-bit A layout
        float v = (k < K) ? row[k] : 0.0f;
        _Float16 h = (_Float16)v;
        hi[i] = h;
        lo[i] = (_Float16)(v - (float)h);
    }
    *((v16h*)APH + t) = hi;
    *((v16h*)APL + t) = lo;
}

__global__ void pack_b_kernel(const float* __restrict__ B, int ldb, int N, int K,
                              int KT, _Float16* __restrict__ BPH,
                              _Float16* __restrict__ BPL)
{
    int t = blockIdx.x * blockDim.x + threadIdx.x;
    int total = (N >> 4) * KT * 32;
    if (t >= total) return;
    const int lane = t & 31;
    const int rem  = t >> 5;
    const int kt   = rem % KT;
    const int nt   = rem / KT;
    const int n    = (nt << 4) + (lane & 15);
    const int kb   = (kt << 5) + ((lane >> 4) << 3);        // B: k sub-base 0/8
    v16h hi, lo;
#pragma unroll
    for (int i = 0; i < 16; ++i) {
        int k = kb + (i & 7) + ((i >> 3) << 4);             // same K layout as A
        float v = (k < K) ? B[(size_t)k * ldb + n] : 0.0f;
        _Float16 h = (_Float16)v;
        hi[i] = h;
        lo[i] = (_Float16)(v - (float)h);
    }
    *((v16h*)BPH + t) = hi;
    *((v16h*)BPL + t) = lo;
}

// ===========================================================================
// GEMM on packed operands: C = act(A@B + bias? + addend?), hi/lo f16 split
// (AhBh + AhBl + AlBh, fp32 accumulate => near-fp32 accuracy).
// Block = 128 threads (4 waves); wave -> 16(M) x 64(N) strip; no guards in loop.
// ===========================================================================
__global__ void __launch_bounds__(128)
gemm_wmma_kernel(const _Float16* __restrict__ APH, const _Float16* __restrict__ APL,
                 const _Float16* __restrict__ BPH, const _Float16* __restrict__ BPL,
                 float* __restrict__ C, int ldc,
                 const float* __restrict__ bias,
                 const float* __restrict__ addend, int ldadd,
                 int M, int N, int KT, int relu)
{
    const int lane = threadIdx.x & 31;
    const int wave = threadIdx.x >> 5;
    const int m0 = blockIdx.y * 64 + wave * 16;
    const int n0 = blockIdx.x * 64;
    if (m0 >= M) return;                        // wave-uniform; EXEC stays full

    const int mt  = m0 >> 4;
    const int nt0 = blockIdx.x * 4;

    const v16h* __restrict__ aH = (const v16h*)APH;
    const v16h* __restrict__ aL = (const v16h*)APL;
    const v16h* __restrict__ bH = (const v16h*)BPH;
    const v16h* __restrict__ bL = (const v16h*)BPL;

    const size_t abase = (size_t)mt * KT * 32 + lane;
    size_t bbase[4];
#pragma unroll
    for (int nf = 0; nf < 4; ++nf) bbase[nf] = (size_t)(nt0 + nf) * KT * 32 + lane;

    v8f acc[4] = {};
    for (int kt = 0; kt < KT; ++kt) {
        const v16h ah = aH[abase + (size_t)kt * 32];
        const v16h al = aL[abase + (size_t)kt * 32];
        if (kt + 1 < KT)
            __builtin_prefetch(&bH[bbase[0] + (size_t)(kt + 1) * 32], 0, 0);
#pragma unroll
        for (int nf = 0; nf < 4; ++nf) {
            const v16h bh = bH[bbase[nf] + (size_t)kt * 32];
            const v16h bl = bL[bbase[nf] + (size_t)kt * 32];
            acc[nf] = __builtin_amdgcn_wmma_f32_16x16x32_f16(
                false, ah, false, bh, (short)0, acc[nf], false, false);
            acc[nf] = __builtin_amdgcn_wmma_f32_16x16x32_f16(
                false, ah, false, bl, (short)0, acc[nf], false, false);
            acc[nf] = __builtin_amdgcn_wmma_f32_16x16x32_f16(
                false, al, false, bh, (short)0, acc[nf], false, false);
        }
    }

    // ---- epilogue: C/D 16x16 f32 layout (VGPR r: M = r + 8*(lane>=16)) ----
    const int mrb = m0 + ((lane >> 4) << 3);
#pragma unroll
    for (int nf = 0; nf < 4; ++nf) {
        const int n = n0 + nf * 16 + (lane & 15);
        const float bv = bias ? bias[n] : 0.0f;
#pragma unroll
        for (int r = 0; r < 8; ++r) {
            const int m = mrb + r;
            float v = acc[nf][r] + bv;
            if (addend) v += addend[(size_t)m * ldadd + n];
            if (relu)   v = fmaxf(v, 0.0f);
            C[(size_t)m * ldc + n] = v;
        }
    }
}

// ---------------------------------------------------------------------------
// Fold attention vector into projection: out[K,H] = sum_d W[k,h*256+d]*attn[h*256+d]
// (makes z_dst unnecessary: er = h @ out, saving a 205 MB tensor)
// ---------------------------------------------------------------------------
__global__ void fold_attn_kernel(const float* __restrict__ W,
                                 const float* __restrict__ attn,
                                 float* __restrict__ out, int K)
{
    int t = blockIdx.x * blockDim.x + threadIdx.x;
    if (t >= K * NHEAD) return;
    int k = t >> 2, h = t & 3;
    const float* wr = W + (size_t)k * HDFLAT + h * HIDN;
    const float* ar = attn + h * HIDN;
    float s = 0.0f;
    for (int d = 0; d < HIDN; ++d) s += wr[d] * ar[d];
    out[t] = s;
}

// out[M,NC] = A[M,K] @ W[K,NC] (+bias[NC]) ; one wave per row, NC <= 4
__global__ void matvec_narrow_kernel(const float* __restrict__ A, int lda,
                                     const float* __restrict__ W,
                                     const float* __restrict__ bias,
                                     float* __restrict__ out, int ldo,
                                     int M, int K, int NC)
{
    int row  = (blockIdx.x * blockDim.x + threadIdx.x) >> 5;
    int lane = threadIdx.x & 31;
    if (row >= M) return;
    const float* ar = A + (size_t)row * lda;
    float acc[4] = {0.f, 0.f, 0.f, 0.f};
    for (int k = lane; k < K; k += 32) {
        float a = ar[k];
        for (int c = 0; c < NC; ++c) acc[c] += a * W[k * NC + c];
    }
    for (int off = 16; off > 0; off >>= 1)
        for (int c = 0; c < NC; ++c) acc[c] += __shfl_down(acc[c], off, 32);
    if (lane == 0)
        for (int c = 0; c < NC; ++c)
            out[(size_t)row * ldo + c] = acc[c] + (bias ? bias[c] : 0.f);
}

// elog[e,h] = leaky_relu( el[src[e],h] + er[dst[e],h] (+ ef[e]*ce[h]) )
__global__ void edge_logit_kernel(const int* __restrict__ src,
                                  const int* __restrict__ dst,
                                  const float* __restrict__ el,
                                  const float* __restrict__ er,
                                  const float* __restrict__ ef,
                                  const float* __restrict__ ce,
                                  float* __restrict__ elog, int E)
{
    int t = blockIdx.x * blockDim.x + threadIdx.x;
    if (t >= E * NHEAD) return;
    int e = t >> 2, h = t & 3;
    float v = el[src[e] * NHEAD + h] + er[dst[e] * NHEAD + h];
    if (ef) v += ef[e] * ce[h];
    elog[t] = (v > 0.f) ? v : 0.2f * v;
}

// ------------------------ CSR build (deterministic agg) ---------------------
__global__ void fill_int_kernel(int* __restrict__ p, int v, int n) {
    int t = blockIdx.x * blockDim.x + threadIdx.x;
    if (t < n) p[t] = v;
}
__global__ void degree_kernel(const int* __restrict__ dst, int* __restrict__ deg, int E) {
    int t = blockIdx.x * blockDim.x + threadIdx.x;
    if (t < E) atomicAdd(&deg[dst[t]], 1);
}
// single-block exclusive scan: rowptr[0..n], cursor[i] = rowptr[i]
__global__ void __launch_bounds__(1024)
scan_kernel(const int* __restrict__ deg, int* __restrict__ rowptr,
            int* __restrict__ cursor, int n)
{
    __shared__ int buf[1024];
    __shared__ int carry;
    if (threadIdx.x == 0) { carry = 0; rowptr[0] = 0; }
    __syncthreads();
    for (int base = 0; base < n; base += 1024) {
        int i = base + threadIdx.x;
        int v = (i < n) ? deg[i] : 0;
        buf[threadIdx.x] = v;
        __syncthreads();
        for (int off = 1; off < 1024; off <<= 1) {
            int t = (threadIdx.x >= off) ? buf[threadIdx.x - off] : 0;
            __syncthreads();
            buf[threadIdx.x] += t;
            __syncthreads();
        }
        int incl = buf[threadIdx.x] + carry;
        if (i < n) { rowptr[i + 1] = incl; cursor[i] = incl - v; }
        __syncthreads();
        if (threadIdx.x == 1023) carry = incl;
        __syncthreads();
    }
}
__global__ void scatter_kernel(const int* __restrict__ dst, int* __restrict__ cursor,
                               int* __restrict__ eidx, int E) {
    int t = blockIdx.x * blockDim.x + threadIdx.x;
    if (t < E) { int p = atomicAdd(&cursor[dst[t]], 1); eidx[p] = t; }
}

// ---------------------------------------------------------------------------
// GAT aggregation, one block per dst node; fuses softmax + weighted sum of
// z_src rows + head-mean.  aggmean[d, 0:256] = mean_h sum_e alpha * z_src.
// ---------------------------------------------------------------------------
__global__ void __launch_bounds__(256)
gat_aggregate_kernel(const int* __restrict__ rowptr, const int* __restrict__ eidx,
                     const int* __restrict__ src, const float* __restrict__ elog,
                     const float* __restrict__ zs, float* __restrict__ aggmean)
{
    __shared__ float sm[NHEAD], sinv[NHEAD];
    __shared__ float accbuf[NHEAD * HIDN];
    const int d   = blockIdx.x;
    const int beg = rowptr[d], end = rowptr[d + 1];

    if (threadIdx.x < NHEAD) {
        const int h = threadIdx.x;
        float mx = -INFINITY;
        for (int j = beg; j < end; ++j) mx = fmaxf(mx, elog[eidx[j] * NHEAD + h]);
        float sum = 0.f;
        for (int j = beg; j < end; ++j) sum += expf(elog[eidx[j] * NHEAD + h] - mx);
        sm[h]   = mx;
        sinv[h] = 1.0f / fmaxf(sum, 1e-9f);
    }
    __syncthreads();

    const int h  = threadIdx.x >> 6;          // 64 threads per head
    const int dq = (threadIdx.x & 63) << 2;   // float4 over D=256
    float4 acc = make_float4(0.f, 0.f, 0.f, 0.f);
    const float mh = sm[h], ih = sinv[h];
    for (int j = beg; j < end; ++j) {
        const int e = eidx[j];
        const float w = expf(elog[e * NHEAD + h] - mh) * ih;
        const float4 z = *reinterpret_cast<const float4*>(
            zs + (size_t)src[e] * HDFLAT + h * HIDN + dq);
        acc.x += w * z.x; acc.y += w * z.y; acc.z += w * z.z; acc.w += w * z.w;
    }
    float* ab = &accbuf[h * HIDN + dq];
    ab[0] = acc.x; ab[1] = acc.y; ab[2] = acc.z; ab[3] = acc.w;
    __syncthreads();

    if (threadIdx.x < 64) {
        const int q = threadIdx.x << 2;
        for (int c = 0; c < 4; ++c) {
            float s = accbuf[q + c] + accbuf[HIDN + q + c] +
                      accbuf[2 * HIDN + q + c] + accbuf[3 * HIDN + q + c];
            aggmean[(size_t)d * HIDN + q + c] = 0.25f * s;
        }
    }
}

// ---------------------------------------------------------------------------
extern "C" void kernel_launch(void* const* d_in, const int* in_sizes, int n_in,
                              void* d_out, int out_size, void* d_ws, size_t ws_size,
                              hipStream_t stream)
{
    // ---- inputs (setup_inputs dict order) ----
    const float* x_link  = (const float*)d_in[0];
    const float* x_flow  = (const float*)d_in[1];
    const float* x_path  = (const float*)d_in[2];
    const float* e2p     = (const float*)d_in[3];
    const float* Wp_link = (const float*)d_in[4];  const float* bp_link = (const float*)d_in[5];
    const float* Wp_flow = (const float*)d_in[6];  const float* bp_flow = (const float*)d_in[7];
    const float* Wp_path = (const float*)d_in[8];  const float* bp_path = (const float*)d_in[9];
    const float* fc_src1 = (const float*)d_in[10]; const float* fc_dst1 = (const float*)d_in[11];
    const float* fc_e1   = (const float*)d_in[12];
    const float* attn_l1 = (const float*)d_in[13]; const float* attn_r1 = (const float*)d_in[14];
    const float* attn_e1 = (const float*)d_in[15];
    const float* res_W1  = (const float*)d_in[16];
    const float* fc_src2 = (const float*)d_in[17]; const float* fc_dst2 = (const float*)d_in[18];
    const float* attn_l2 = (const float*)d_in[19]; const float* attn_r2 = (const float*)d_in[20];
    const float* res_W2  = (const float*)d_in[21];
    const float* W1      = (const float*)d_in[22]; const float* b1 = (const float*)d_in[23];
    const float* W2      = (const float*)d_in[24]; const float* b2 = (const float*)d_in[25];
    const int*   src1    = (const int*)d_in[26];   const int* dst1 = (const int*)d_in[27];
    const int*   src2    = (const int*)d_in[28];   const int* dst2 = (const int*)d_in[29];
    float* out = (float*)d_out;

    // ---- workspace carve (bump allocator, 256B aligned) ----
    char* base = (char*)d_ws; size_t off = 0;
    auto carve = [&](size_t bytes) -> void* {
        void* p = base + off; off = (off + bytes + 255) & ~(size_t)255; return p;
    };
    float* HL   = (float*)carve((size_t)N_LINK * HIDN * 4);
    float* HF   = (float*)carve((size_t)N_FLOW * HIDN * 4);
    float* HP0  = (float*)carve((size_t)N_PATH * HIDN * 4);  // later reused as decoder T
    float* ZS1  = (float*)carve((size_t)N_LINK * HDFLAT * 4);
    float* ZS2  = (float*)carve((size_t)N_FLOW * HDFLAT * 4);
    float* P1   = (float*)carve((size_t)N_PATH * HIDN * 4);
    float* P2   = (float*)carve((size_t)N_PATH * HIDN * 4);
    float* AGG  = (float*)carve((size_t)N_PATH * HIDN * 4);  // later reused as HDEC
    float* EL1  = (float*)carve((size_t)N_LINK * NHEAD * 4);
    float* ER1  = (float*)carve((size_t)N_PATH * NHEAD * 4);
    float* EL2  = (float*)carve((size_t)N_FLOW * NHEAD * 4);
    float* ER2  = (float*)carve((size_t)N_PATH * NHEAD * 4);
    float* WL1  = (float*)carve(HIDN * NHEAD * 4);
    float* WR1  = (float*)carve(HIDN * NHEAD * 4);
    float* WL2  = (float*)carve(HIDN * NHEAD * 4);
    float* WR2  = (float*)carve(HIDN * NHEAD * 4);
    float* CE   = (float*)carve(NHEAD * 4);
    float* ELOG = (float*)carve((size_t)N_EDGE * NHEAD * 4);
    int*   deg    = (int*)carve((size_t)N_PATH * 4);
    int*   rowptr = (int*)carve((size_t)(N_PATH + 1) * 4);
    int*   cursor = (int*)carve((size_t)N_PATH * 4);
    int*   eidx   = (int*)carve((size_t)N_EDGE * 4);
    // packed operand planes (max A: 50000x256; max B: 256x1024)
    const size_t APK_MAX = (size_t)N_PATH * HIDN;            // halves per plane
    const size_t BPK_MAX = (size_t)HIDN * HDFLAT;
    _Float16* APH = (_Float16*)carve(APK_MAX * 2);
    _Float16* APL = (_Float16*)carve(APK_MAX * 2);
    _Float16* BPH = (_Float16*)carve(BPK_MAX * 2);
    _Float16* BPL = (_Float16*)carve(BPK_MAX * 2);

    auto gemm = [&](const float* A, int lda, const float* B, int ldb,
                    float* C, int ldc, const float* bias,
                    const float* addend, int ldadd, int M, int N, int K, int relu) {
        const int KT = (K + 31) >> 5;
        const int na = (M >> 4) * KT * 32;
        const int nb = (N >> 4) * KT * 32;
        pack_a_kernel<<<(na + 255) / 256, 256, 0, stream>>>(A, lda, M, K, KT, APH, APL);
        pack_b_kernel<<<(nb + 255) / 256, 256, 0, stream>>>(B, ldb, N, K, KT, BPH, BPL);
        dim3 g(N / 64, (M + 63) / 64);
        gemm_wmma_kernel<<<g, 128, 0, stream>>>(APH, APL, BPH, BPL, C, ldc,
                                                bias, addend, ldadd, M, N, KT, relu);
    };
    auto csr = [&](const int* dst) {
        fill_int_kernel<<<(N_PATH + 255) / 256, 256, 0, stream>>>(deg, 0, N_PATH);
        degree_kernel<<<(N_EDGE + 255) / 256, 256, 0, stream>>>(dst, deg, N_EDGE);
        scan_kernel<<<1, 1024, 0, stream>>>(deg, rowptr, cursor, N_PATH);
        scatter_kernel<<<(N_EDGE + 255) / 256, 256, 0, stream>>>(dst, cursor, eidx, N_EDGE);
    };

    // ---- per-type projections (WMMA, K zero-padded to 32 by the packer) ----
    gemm(x_link, 8,  Wp_link, HIDN, HL,  HIDN, bp_link, nullptr, 0, N_LINK, HIDN, 8,  1);
    gemm(x_flow, 16, Wp_flow, HIDN, HF,  HIDN, bp_flow, nullptr, 0, N_FLOW, HIDN, 16, 1);
    gemm(x_path, 8,  Wp_path, HIDN, HP0, HIDN, bp_path, nullptr, 0, N_PATH, HIDN, 8,  1);

    // ---- conv_1: link -> path ----
    gemm(HL, HIDN, fc_src1, HDFLAT, ZS1, HDFLAT, nullptr, nullptr, 0, N_LINK, HDFLAT, HIDN, 0);
    fold_attn_kernel<<<(HIDN * NHEAD + 255) / 256, 256, 0, stream>>>(fc_src1, attn_l1, WL1, HIDN);
    fold_attn_kernel<<<(HIDN * NHEAD + 255) / 256, 256, 0, stream>>>(fc_dst1, attn_r1, WR1, HIDN);
    fold_attn_kernel<<<1, 256, 0, stream>>>(fc_e1, attn_e1, CE, 1);
    matvec_narrow_kernel<<<(N_LINK * 32 + 255) / 256, 256, 0, stream>>>(
        HL, HIDN, WL1, nullptr, EL1, NHEAD, N_LINK, HIDN, NHEAD);
    matvec_narrow_kernel<<<(N_PATH * 32 + 255) / 256, 256, 0, stream>>>(
        HP0, HIDN, WR1, nullptr, ER1, NHEAD, N_PATH, HIDN, NHEAD);
    edge_logit_kernel<<<(N_EDGE * NHEAD + 255) / 256, 256, 0, stream>>>(
        src1, dst1, EL1, ER1, e2p, CE, ELOG, N_EDGE);
    csr(dst1);
    gat_aggregate_kernel<<<N_PATH, 256, 0, stream>>>(rowptr, eidx, src1, ELOG, ZS1, AGG);
    // P1 = relu(agg_mean + HP0 @ res_W1)
    gemm(HP0, HIDN, res_W1, HIDN, P1, HIDN, nullptr, AGG, HIDN, N_PATH, HIDN, HIDN, 1);

    // ---- conv_2: flow -> path ----
    gemm(HF, HIDN, fc_src2, HDFLAT, ZS2, HDFLAT, nullptr, nullptr, 0, N_FLOW, HDFLAT, HIDN, 0);
    fold_attn_kernel<<<(HIDN * NHEAD + 255) / 256, 256, 0, stream>>>(fc_src2, attn_l2, WL2, HIDN);
    fold_attn_kernel<<<(HIDN * NHEAD + 255) / 256, 256, 0, stream>>>(fc_dst2, attn_r2, WR2, HIDN);
    matvec_narrow_kernel<<<(N_FLOW * 32 + 255) / 256, 256, 0, stream>>>(
        HF, HIDN, WL2, nullptr, EL2, NHEAD, N_FLOW, HIDN, NHEAD);
    matvec_narrow_kernel<<<(N_PATH * 32 + 255) / 256, 256, 0, stream>>>(
        P1, HIDN, WR2, nullptr, ER2, NHEAD, N_PATH, HIDN, NHEAD);
    edge_logit_kernel<<<(N_EDGE * NHEAD + 255) / 256, 256, 0, stream>>>(
        src2, dst2, EL2, ER2, nullptr, nullptr, ELOG, N_EDGE);
    csr(dst2);
    gat_aggregate_kernel<<<N_PATH, 256, 0, stream>>>(rowptr, eidx, src2, ELOG, ZS2, AGG);
    // P2 = relu(agg_mean + P1 @ res_W2)
    gemm(P1, HIDN, res_W2, HIDN, P2, HIDN, nullptr, AGG, HIDN, N_PATH, HIDN, HIDN, 1);

    // ---- decoder: concat([P2, P1, P2]) @ W1 done as 3 accumulating GEMMs ----
    float* T    = HP0;  // HP0 dead from here
    float* HDEC = AGG;  // AGG dead after conv2 residual
    gemm(P1, HIDN, W1 + 256 * 256, HIDN, T, HIDN, nullptr, nullptr, 0, N_PATH, HIDN, HIDN, 0);
    gemm(P2, HIDN, W1,             HIDN, T, HIDN, nullptr, T, HIDN, N_PATH, HIDN, HIDN, 0);
    gemm(P2, HIDN, W1 + 512 * 256, HIDN, HDEC, HIDN, b1, T, HIDN, N_PATH, HIDN, HIDN, 1);

    // ---- final linear: out[50000,2] = HDEC @ W2 + b2 ----
    matvec_narrow_kernel<<<(N_PATH * 32 + 255) / 256, 256, 0, stream>>>(
        HDEC, HIDN, W2, b2, out, 2, N_PATH, HIDN, 2);
}
